// LSTM_12386685681854
// MI455X (gfx1250) — matrix operand order repaired and checked by reading, not compiled, not run on hardware
//
#include <hip/hip_runtime.h>
#include <stdint.h>

#define T_SEQ   512
#define D_IN    64
#define H_DIM   128
#define N_CLS   10
#define N_BATCH 1024
#define N_GATES 512   // 4*H

// ---- fragment / vector types -------------------------------------------------
typedef __attribute__((ext_vector_type(16))) __bf16       v16bf;
typedef __attribute__((ext_vector_type(2)))  __bf16       v2bf;
typedef __attribute__((ext_vector_type(8)))  float        v8f;
typedef __attribute__((ext_vector_type(4)))  unsigned int u32x4;
typedef __attribute__((ext_vector_type(2)))  unsigned int u32x2;
typedef __attribute__((ext_vector_type(4)))  float        f32x4;

struct Frag32B { u32x4 lo, hi; };

static __device__ __forceinline__ unsigned short f2bf(float f) {
  unsigned u = __builtin_bit_cast(unsigned, f);
  u += 0x7fffu + ((u >> 16) & 1u);          // round-to-nearest-even
  return (unsigned short)(u >> 16);
}

// fp32 pair -> packed bf16 (one v_cvt_pk_bf16_f32 when available)
static __device__ __forceinline__ unsigned pack_bf16(float a, float b) {
#if __has_builtin(__builtin_amdgcn_cvt_pk_bf16_f32)
  v2bf r = __builtin_amdgcn_cvt_pk_bf16_f32(a, b);
  return __builtin_bit_cast(unsigned, r);
#else
  return (unsigned)f2bf(a) | ((unsigned)f2bf(b) << 16);
#endif
}

// Load a 32-byte WMMA fragment as two 16-byte pieces separated by `gap` bytes.
// A-fragments (16x32 bf16): gap = 16.  B-fragments (32x16 bf16): gap = 16.
static __device__ __forceinline__ v16bf load_frag(const char* p, int gap) {
  Frag32B t;
  t.lo = *(const u32x4*)(p);
  t.hi = *(const u32x4*)(p + gap);
  return __builtin_bit_cast(v16bf, t);
}

// Hardware tanh (v_tanh_f32 is a gfx1250 TRANS op -> co-executes with WMMA).
static __device__ __forceinline__ float tanh_fast(float x) {
#if __has_builtin(__builtin_amdgcn_tanhf)
  return __builtin_amdgcn_tanhf(x);
#elif __has_builtin(__builtin_amdgcn_rcpf)
  return 1.0f - 2.0f * __builtin_amdgcn_rcpf(__expf(2.0f * x) + 1.0f);
#else
  return 1.0f - 2.0f / (__expf(2.0f * x) + 1.0f);
#endif
}
static __device__ __forceinline__ float sig_fast(float x) {
  return 0.5f + 0.5f * tanh_fast(0.5f * x);   // sigmoid via tanh
}

// ---- prologue: pack weights to bf16, N-major ([n_global][k]) -----------------
// n_global = gate*128 + n, gate order: g, i, f, o.
__global__ void lstm_pack(const float* Wgx, const float* Wgh, const float* Wix,
                          const float* Wih, const float* Wfx, const float* Wfh,
                          const float* Wox, const float* Woh,
                          const float* bg, const float* bi, const float* bf_,
                          const float* bo,
                          unsigned short* wxT, unsigned short* whT, float* ball) {
  const int tid = blockIdx.x * blockDim.x + threadIdx.x;
  const int nth = gridDim.x * blockDim.x;
  const float* Wx[4] = {Wgx, Wix, Wfx, Wox};
  const float* Wh[4] = {Wgh, Wih, Wfh, Woh};
  const float* bb[4] = {bg, bi, bf_, bo};

  for (int i = tid; i < N_GATES * D_IN; i += nth) {       // wxT[n][k], k fast
    int n = i / D_IN, k = i - n * D_IN;
    wxT[i] = f2bf(Wx[n >> 7][k * H_DIM + (n & 127)]);
  }
  for (int i = tid; i < N_GATES * H_DIM; i += nth) {      // whT[n][k], k fast
    int n = i / H_DIM, k = i - n * H_DIM;
    whT[i] = f2bf(Wh[n >> 7][k * H_DIM + (n & 127)]);
  }
  for (int i = tid; i < N_GATES; i += nth) ball[i] = bb[i >> 7][i & 127];
}

// ---- LDS layout (bytes) ------------------------------------------------------
#define WH_STRIDE_B 272                    // 136 bf16: 16B-aligned, 68 dwords (bank-safe)
#define WH_BYTES    (N_GATES * WH_STRIDE_B)        // 139264
#define HB_OFF      WH_BYTES
#define HB_BYTES    (16 * WH_STRIDE_B)             // 4352
#define XB_OFF      (HB_OFF + HB_BYTES)            // 143616
#define XB_STRIDE_B 176                    // 88 bf16: 16B-aligned, 44 dwords (bank-safe)
#define XB_BYTES    (16 * XB_STRIDE_B)             // 2816
#define SMEM_BYTES  (XB_OFF + XB_BYTES)            // 146432  (< 160KB CU-mode half)

// ---- main persistent LSTM kernel --------------------------------------------
__global__ __launch_bounds__(256)
void lstm_main(const float* __restrict__ x, const unsigned short* __restrict__ wxT,
               const unsigned short* __restrict__ whT, const float* __restrict__ ball,
               const float* __restrict__ Wph, const float* __restrict__ bp,
               float* __restrict__ out) {
  extern __shared__ char smem[];
  char* shWh = smem;            // Wh_all^T, padded, bf16
  char* shH  = smem + HB_OFF;   // h tile [16][136] bf16
  char* shX  = smem + XB_OFF;   // x_t tile [16][88] bf16

  const int tid    = threadIdx.x;
  const int lane   = tid & 31;
  const int wv     = tid >> 5;            // 8 waves
  const int laneLo = lane & 15;
  const int laneHi = lane >> 4;
  const int hs     = wv * 16;             // hidden slice owned by this wave
  const int m0     = blockIdx.x * 16;     // batch-row tile

  // Stage Wh_all^T into padded LDS (one-time; 128KB).
  for (int c = tid; c < N_GATES * 16; c += 256) {
    int row = c >> 4, kc = (c & 15) * 8;
    *(u32x4*)(shWh + row * WH_STRIDE_B + kc * 2) =
        *(const u32x4*)((const char*)whT + (row * H_DIM + kc) * 2);
  }
  // h0 = 0 (zero pad too).
  for (int i = tid * 8; i < 16 * 136; i += 256 * 8)
    *(u32x4*)(shH + i * 2) = (u32x4){0u, 0u, 0u, 0u};

  // Per-lane gate biases (column n = hs+laneLo, same for every row of C;
  // the splat C-registers are loop-invariant -> hoisted by the compiler).
  float bias[4];
#pragma unroll
  for (int g = 0; g < 4; ++g) bias[g] = ball[g * H_DIM + hs + laneLo];

  v8f c = {};                             // cell-state slice, lives in VGPRs

  const size_t xrow = (size_t)T_SEQ * D_IN;
  const int xm = tid >> 4, xd = (tid & 15) * 4;           // x staging coords
  const float* xbase = x + (size_t)(m0 + xm) * xrow + xd;

  __syncthreads();

  for (int t = 0; t < T_SEQ; ++t) {
    // ---- stage x_t -> LDS as bf16, prefetch x_{t+1} (the HBM stream) ----
    {
      f32x4 v = *(const f32x4*)(xbase + (size_t)t * D_IN);
      if ((tid & 7) == 0 && t + 1 < T_SEQ)       // one prefetch per 128B line
        __builtin_prefetch(xbase + (size_t)(t + 1) * D_IN, 0, 1);
      u32x2 p;
      p.x = pack_bf16(v.x, v.y);
      p.y = pack_bf16(v.z, v.w);
      *(u32x2*)(shX + xm * XB_STRIDE_B + xd * 2) = p;
    }
    __syncthreads();   // x_t staged; previous step's h writes visible

    // ---- A fragments (shared across the 4 gates) ----
    v16bf ax[2], ah[4];
#pragma unroll
    for (int kt = 0; kt < 2; ++kt)
      ax[kt] = load_frag(shX + laneLo * XB_STRIDE_B + kt * 64 + laneHi * 32, 16);
#pragma unroll
    for (int kt = 0; kt < 4; ++kt)
      ah[kt] = load_frag(shH + laneLo * WH_STRIDE_B + kt * 64 + laneHi * 32, 16);

    // ---- gates: 24 WMMAs (4 gates x (2 + 4) K-tiles) ----
    v8f acc[4];
#pragma unroll
    for (int g = 0; g < 4; ++g) {
      float b0 = bias[g];
      v8f a = {b0, b0, b0, b0, b0, b0, b0, b0};
      const int ng = g * H_DIM + hs + laneLo;             // B column (N index)
      const char* wxb = (const char*)wxT + ng * (D_IN * 2) + laneHi * 32;
#pragma unroll
      for (int kt = 0; kt < 2; ++kt) {
        v16bf bfr = load_frag(wxb + kt * 64, 16);         // loop-invariant -> regs
        a = __builtin_amdgcn_wmma_f32_16x16x32_bf16(false, ax[kt], false, bfr,
                                                    (short)0, a, false, false);
      }
      const char* whb = shWh + ng * WH_STRIDE_B + laneHi * 32;
#pragma unroll
      for (int kt = 0; kt < 4; ++kt) {
        v16bf bfr = load_frag(whb + kt * 64, 16);
        a = __builtin_amdgcn_wmma_f32_16x16x32_bf16(false, ah[kt], false, bfr,
                                                    (short)0, a, false, false);
      }
      acc[g] = a;
    }
    __syncthreads();   // all LDS reads of shH/shX done before overwrite

    // ---- pointwise state update (TRANS-heavy, co-executes with WMMA) ----
#pragma unroll
    for (int r = 0; r < 8; ++r) {
      float gv = tanh_fast(acc[0][r]);
      float iv = sig_fast(acc[1][r]);
      float fv = sig_fast(acc[2][r]);
      float ov = sig_fast(acc[3][r]);
      float cv = gv * iv + c[r] * fv;
      c[r] = cv;
      float hv = tanh_fast(cv) * ov;
      int row = r + 8 * laneHi;                           // C/D layout row
      *(unsigned short*)(shH + row * WH_STRIDE_B + (hs + laneLo) * 2) =
          (unsigned short)pack_bf16(hv, 0.0f);
    }
  }
  __syncthreads();

  // ---- output projection: [16 x 128] @ Wph[128 x 10] + bp (reads bf16 h) ----
  if (tid < 16 * N_CLS) {
    int m = tid / N_CLS, cls = tid - m * N_CLS;
    const unsigned short* hrow = (const unsigned short*)(shH + m * WH_STRIDE_B);
    float s = bp[cls];
    for (int k = 0; k < H_DIM; ++k) {
      float hk = __builtin_bit_cast(float, (unsigned)hrow[k] << 16);
      s += hk * Wph[k * N_CLS + cls];
    }
    out[(m0 + m) * N_CLS + cls] = s;
  }
}

// ---- launch ------------------------------------------------------------------
extern "C" void kernel_launch(void* const* d_in, const int* in_sizes, int n_in,
                              void* d_out, int out_size, void* d_ws, size_t ws_size,
                              hipStream_t stream) {
  const float* x   = (const float*)d_in[0];
  const float* Wgx = (const float*)d_in[1];
  const float* Wgh = (const float*)d_in[2];
  const float* Wix = (const float*)d_in[3];
  const float* Wih = (const float*)d_in[4];
  const float* Wfx = (const float*)d_in[5];
  const float* Wfh = (const float*)d_in[6];
  const float* Wox = (const float*)d_in[7];
  const float* Woh = (const float*)d_in[8];
  const float* Wph = (const float*)d_in[9];
  const float* bg  = (const float*)d_in[10];
  const float* bi  = (const float*)d_in[11];
  const float* bf  = (const float*)d_in[12];
  const float* bo  = (const float*)d_in[13];
  const float* bp  = (const float*)d_in[14];

  // workspace: wxT 64KB | whT 128KB | ball 2KB  (~195KB total)
  unsigned short* wxT = (unsigned short*)d_ws;
  unsigned short* whT = wxT + N_GATES * D_IN;
  float*          ball = (float*)(whT + N_GATES * H_DIM);

  lstm_pack<<<64, 256, 0, stream>>>(Wgx, Wgh, Wix, Wih, Wfx, Wfh, Wox, Woh,
                                    bg, bi, bf, bo, wxT, whT, ball);

  (void)hipFuncSetAttribute((const void*)lstm_main,
                            hipFuncAttributeMaxDynamicSharedMemorySize, SMEM_BYTES);
  lstm_main<<<N_BATCH / 16, 256, SMEM_BYTES, stream>>>(x, wxT, whT, ball, Wph, bp,
                                                       (float*)d_out);
}
